// MultiEdgeGraphBlock_42691974922272
// MI455X (gfx1250) — matrix operand order repaired and checked by reading, not compiled, not run on hardware
//
#include <hip/hip_runtime.h>
#include <hip/hip_bf16.h>

// Problem constants (match reference)
#define BB 4
#define NN 10000
#define FF 256
#define HH 256
#define EE 5
#define DG 16
#define XD (FF + HH)   // 512
#define LN_EPS 1e-6f

typedef __attribute__((ext_vector_type(16))) __bf16 v16bf;
typedef __attribute__((ext_vector_type(8)))  float  v8f;
typedef unsigned int u32;

__device__ __forceinline__ v8f zero8() {
  v8f z = {0.f, 0.f, 0.f, 0.f, 0.f, 0.f, 0.f, 0.f};
  return z;
}

// A fragment (16x32 bf16) from an LDS row-major tile.
// CDNA5 layout: lane(0-15)=row M, lanes 16-31 same rows with K offset +8;
// VGPR j<4 holds K=2j,2j+1 (+khalf), VGPR j>=4 holds K=16+2(j-4),+1 (+khalf).
__device__ __forceinline__ v16bf a_frag_lds(const __bf16* sA, int ld, int lane, int k0) {
  int row   = lane & 15;
  int khalf = (lane & 16) ? 8 : 0;
  v16bf a;
#pragma unroll
  for (int j = 0; j < 8; ++j) {
    int kb = k0 + khalf + ((j < 4) ? (2 * j) : (16 + 2 * (j - 4)));
    a[2 * j]     = sA[row * ld + kb];
    a[2 * j + 1] = sA[row * ld + kb + 1];
  }
  return a;
}

// B fragment from pre-swizzled packed-bf16 weights: fragment `frag` holds
// 32 lanes x 8 dwords contiguously, already in B-matrix register order.
// Two coalesced 128-bit loads per fragment.
__device__ __forceinline__ v16bf b_frag_sw(const u32* __restrict__ Wsw, int frag, int lane) {
  const uint4* p = (const uint4*)(Wsw + (((size_t)frag * 32 + lane) << 3));
  union { uint4 q[2]; v16bf v; } u;
  u.q[0] = p[0];
  u.q[1] = p[1];
  return u.v;
}

// Pre-swizzle an f32 row-major weight [K][N] into packed-bf16 B fragments.
// dst[((kt*nct + ct)*32 + lane)*8 + j] packs (k, k+1) for
// col = ct*16 + (lane&15), k = kt*32 + (lane&16 ? 8 : 0) + (j<4 ? 2j : 16+2(j-4)).
__global__ __launch_bounds__(256) void swizzle_weights(
    const float* __restrict__ W, u32* __restrict__ dst, int K, int N) {
  int idx = blockIdx.x * 256 + threadIdx.x;
  if (idx >= (K * N) / 2) return;
  int j    = idx & 7;
  int lane = (idx >> 3) & 31;
  int frag = idx >> 8;
  int nct  = N >> 4;
  int ct   = frag % nct;
  int kt   = frag / nct;
  int col  = ct * 16 + (lane & 15);
  int k    = kt * 32 + ((lane & 16) ? 8 : 0) + ((j < 4) ? (2 * j) : (16 + 2 * (j - 4)));
  union { __bf16 h[2]; u32 u; } p;
  p.h[0] = (__bf16)W[(size_t)k * N + col];
  p.h[1] = (__bf16)W[(size_t)(k + 1) * N + col];
  dst[idx] = p.u;
}

__global__ __launch_bounds__(256) void fused_graph_block(
    const float* __restrict__ hin,
    const int*   __restrict__ eidx,
    const float* __restrict__ emask,
    const u32*   __restrict__ wedge_sw,   // [E*8 ktiles][16 ct] frags
    const float* __restrict__ b_edge,
    const float* __restrict__ ln_scale,
    const float* __restrict__ ln_bias,
    const u32*   __restrict__ w1_sw,      // [16 ktiles][16 ct] frags
    const float* __restrict__ b1,
    const u32*   __restrict__ w2_sw,      // [8 ktiles][16 ct] frags
    const float* __restrict__ b2,
    float* __restrict__ out)
{
  __shared__ float  sXf[16][XD];        // concat(h, aggregated) fp32   (32 KB)
  __shared__ __bf16 sXn[16][XD + 8];    // normalized bf16              (16.25 KB)
  __shared__ __bf16 sHid[16][HH + 8];   // relu(x@W1+b1) bf16           (8.25 KB)
  // gather tile aliases sXn's storage (dead before sXn is written; barriers separate)
  __bf16 (*sA)[FF + 8] = (__bf16(*)[FF + 8]) & sXn[0][0];

  const int tid  = threadIdx.x;
  const int lane = tid & 31;
  const int wave = tid >> 5;
  const int b    = blockIdx.y;
  const int n0   = blockIdx.x * 16;

  const int ct0    = wave * 2;           // this wave's two 16-wide column tiles
  const int col0   = ct0 * 16;
  const int cA     = col0 + (lane & 15);
  const int cB     = cA + 16;
  const int rowoff = (lane & 16) ? 8 : 0;

  // gather mapping: 256 threads = 16 nodes x 16 feature-chunks of 16
  const int r  = tid >> 4;
  const int fb = (tid & 15) * 16;
  const int n  = n0 + r;
  const float* hb = hin + (size_t)b * NN * FF;

  // ---- Phase 1: per-edge masked-mean gather + WMMA accumulate into `aggregated` ----
  v8f agg0 = zero8(), agg1 = zero8();
  for (int e = 0; e < EE; ++e) {
    float accf[16];
#pragma unroll
    for (int i = 0; i < 16; ++i) accf[i] = 0.f;
    float denom = 0.f;
    const int*   ei = eidx  + ((size_t)e * NN + n) * DG;
    const float* em = emask + ((size_t)e * NN + n) * DG;
    for (int d = 0; d < DG; ++d) {
      int idx = ei[d];
      if (idx < 0) idx = 0;
      float m = em[d];
      denom += m;
      if (m != 0.f) {                       // ~half the gather traffic skipped
        const float* src = hb + (size_t)idx * FF + fb;
#pragma unroll
        for (int q = 0; q < 4; ++q) {
          float4 v = ((const float4*)src)[q];
          accf[4 * q + 0] += m * v.x;
          accf[4 * q + 1] += m * v.y;
          accf[4 * q + 2] += m * v.z;
          accf[4 * q + 3] += m * v.w;
        }
      }
    }
    float inv = 1.f / fmaxf(denom, 1.f);
    {
      u32* dstrow = (u32*)&sA[r][fb];
#pragma unroll
      for (int i = 0; i < 8; ++i) {
        union { __bf16 h[2]; u32 u; } p;
        p.h[0] = (__bf16)(accf[2 * i]     * inv);
        p.h[1] = (__bf16)(accf[2 * i + 1] * inv);
        dstrow[i] = p.u;
      }
    }
    __syncthreads();

#pragma unroll
    for (int kt = 0; kt < FF / 32; ++kt) {
      int fbase = (e * (FF / 32) + kt) * 16;
      v16bf a   = a_frag_lds(&sA[0][0], FF + 8, lane, kt * 32);
      v16bf bf0 = b_frag_sw(wedge_sw, fbase + ct0,     lane);
      v16bf bf1 = b_frag_sw(wedge_sw, fbase + ct0 + 1, lane);
      agg0 = __builtin_amdgcn_wmma_f32_16x16x32_bf16(false, a, false, bf0, (short)0, agg0, false, false);
      agg1 = __builtin_amdgcn_wmma_f32_16x16x32_bf16(false, a, false, bf1, (short)0, agg1, false, false);
    }
    __syncthreads();
  }

  // edge biases (summed over e) + write aggregated into sXf[:, FF:]
  float biasA = 0.f, biasB = 0.f;
  for (int e = 0; e < EE; ++e) {
    biasA += b_edge[e * HH + cA];
    biasB += b_edge[e * HH + cB];
  }
#pragma unroll
  for (int v = 0; v < 8; ++v) {
    sXf[v + rowoff][FF + cA] = agg0[v] + biasA;
    sXf[v + rowoff][FF + cB] = agg1[v] + biasB;
  }
  // copy h tile into sXf[:, 0:FF]
  {
    const float* hrow = hb + (size_t)n * FF + fb;
#pragma unroll
    for (int q = 0; q < 4; ++q) {
      float4 v = ((const float4*)hrow)[q];
      sXf[r][fb + 4 * q + 0] = v.x;
      sXf[r][fb + 4 * q + 1] = v.y;
      sXf[r][fb + 4 * q + 2] = v.z;
      sXf[r][fb + 4 * q + 3] = v.w;
    }
  }
  __syncthreads();

  // ---- Phase 2: LayerNorm over 512 (one wave per 2 rows; lane owns 16 contiguous) ----
  for (int rr = 0; rr < 2; ++rr) {
    int row = wave * 2 + rr;
    int c0  = lane * 16;
    float v[16];
#pragma unroll
    for (int q = 0; q < 4; ++q) {
      float4 t = ((const float4*)&sXf[row][c0])[q];
      v[4 * q + 0] = t.x; v[4 * q + 1] = t.y; v[4 * q + 2] = t.z; v[4 * q + 3] = t.w;
    }
    float s = 0.f;
#pragma unroll
    for (int i = 0; i < 16; ++i) s += v[i];
#pragma unroll
    for (int o = 16; o > 0; o >>= 1) s += __shfl_xor(s, o, 32);
    float mu = s * (1.f / XD);
    float vs = 0.f;
#pragma unroll
    for (int i = 0; i < 16; ++i) { float d = v[i] - mu; vs += d * d; }
#pragma unroll
    for (int o = 16; o > 0; o >>= 1) vs += __shfl_xor(vs, o, 32);
    float rs = rsqrtf(vs * (1.f / XD) + LN_EPS);
    u32* dstrow = (u32*)&sXn[row][c0];
#pragma unroll
    for (int i = 0; i < 8; ++i) {
      int c = c0 + 2 * i;
      union { __bf16 h[2]; u32 u; } p;
      p.h[0] = (__bf16)((v[2 * i]     - mu) * rs * ln_scale[c]     + ln_bias[c]);
      p.h[1] = (__bf16)((v[2 * i + 1] - mu) * rs * ln_scale[c + 1] + ln_bias[c + 1]);
      dstrow[i] = p.u;
    }
  }
  __syncthreads();

  // ---- Phase 3: hidden = relu(xn @ W1 + b1), K = 512 ----
  {
    v8f h0 = zero8(), h1 = zero8();
#pragma unroll
    for (int kt = 0; kt < XD / 32; ++kt) {
      v16bf a   = a_frag_lds(&sXn[0][0], XD + 8, lane, kt * 32);
      v16bf bf0 = b_frag_sw(w1_sw, kt * 16 + ct0,     lane);
      v16bf bf1 = b_frag_sw(w1_sw, kt * 16 + ct0 + 1, lane);
      h0 = __builtin_amdgcn_wmma_f32_16x16x32_bf16(false, a, false, bf0, (short)0, h0, false, false);
      h1 = __builtin_amdgcn_wmma_f32_16x16x32_bf16(false, a, false, bf1, (short)0, h1, false, false);
    }
    float bA = b1[cA], bB = b1[cB];
#pragma unroll
    for (int v = 0; v < 8; ++v) {
      sHid[v + rowoff][cA] = (__bf16)fmaxf(h0[v] + bA, 0.f);
      sHid[v + rowoff][cB] = (__bf16)fmaxf(h1[v] + bB, 0.f);
    }
  }
  __syncthreads();

  // ---- Phase 4: out = h + hidden @ W2 + b2, K = 256 ----
  {
    v8f o0 = zero8(), o1 = zero8();
#pragma unroll
    for (int kt = 0; kt < HH / 32; ++kt) {
      v16bf a   = a_frag_lds(&sHid[0][0], HH + 8, lane, kt * 32);
      v16bf bf0 = b_frag_sw(w2_sw, kt * 16 + ct0,     lane);
      v16bf bf1 = b_frag_sw(w2_sw, kt * 16 + ct0 + 1, lane);
      o0 = __builtin_amdgcn_wmma_f32_16x16x32_bf16(false, a, false, bf0, (short)0, o0, false, false);
      o1 = __builtin_amdgcn_wmma_f32_16x16x32_bf16(false, a, false, bf1, (short)0, o1, false, false);
    }
    float bA = b2[cA], bB = b2[cB];
#pragma unroll
    for (int v = 0; v < 8; ++v) {
      size_t off = ((size_t)b * NN + n0 + v + rowoff) * FF;
      out[off + cA] = hin[off + cA] + o0[v] + bA;
      out[off + cB] = hin[off + cB] + o1[v] + bB;
    }
  }
}

extern "C" void kernel_launch(void* const* d_in, const int* in_sizes, int n_in,
                              void* d_out, int out_size, void* d_ws, size_t ws_size,
                              hipStream_t stream) {
  (void)in_sizes; (void)n_in; (void)ws_size; (void)out_size;
  const float* hin    = (const float*)d_in[0];
  const int*   eidx   = (const int*)d_in[1];
  const float* emask  = (const float*)d_in[2];
  const float* W_edge = (const float*)d_in[3];
  const float* b_edge = (const float*)d_in[4];
  const float* ln_s   = (const float*)d_in[5];
  const float* ln_b   = (const float*)d_in[6];
  const float* W1     = (const float*)d_in[7];
  const float* b1     = (const float*)d_in[8];
  const float* W2     = (const float*)d_in[9];
  const float* b2     = (const float*)d_in[10];
  float* out = (float*)d_out;

  // Workspace: packed-bf16 swizzled weights (1 MB total, L2-resident)
  u32* wedge_sw = (u32*)d_ws;                         // 5*256*256/2 = 163840 dwords
  u32* w1_sw    = wedge_sw + (EE * FF * HH) / 2;      // 512*256/2   =  65536 dwords
  u32* w2_sw    = w1_sw + (XD * HH) / 2;              // 256*256/2   =  32768 dwords

  // Pre-swizzle weights (W_edge treated as a single [E*F][H] matrix: fragment
  // order (e*8 + kt)*16 + ct matches the main kernel's indexing).
  swizzle_weights<<<dim3((EE * FF * HH / 2 + 255) / 256), dim3(256), 0, stream>>>(
      W_edge, wedge_sw, EE * FF, HH);
  swizzle_weights<<<dim3((XD * HH / 2 + 255) / 256), dim3(256), 0, stream>>>(
      W1, w1_sw, XD, HH);
  swizzle_weights<<<dim3((HH * HH / 2 + 255) / 256), dim3(256), 0, stream>>>(
      W2, w2_sw, HH, HH);

  dim3 grid(NN / 16, BB, 1);   // 625 node-tiles x 4 batches
  fused_graph_block<<<grid, dim3(256, 1, 1), 0, stream>>>(
      hin, eidx, emask, wedge_sw, b_edge, ln_s, ln_b, w1_sw, b1, w2_sw, b2, out);
}
